// VarAttention_180388626453
// MI455X (gfx1250) — compile-verified, matrix operand order checked by reading, not run on hardware
//
#include <hip/hip_runtime.h>
#include <hip/hip_bf16.h>

// ---------------------------------------------------------------------------
// Problem constants (from reference): b=16, f=8, p=196, dim=768, H=12
// ---------------------------------------------------------------------------
#define BATCH   16
#define HEADS   12
#define FRAMES  8
#define PTOK    196
#define NTOK    1569          // 1 + FRAMES*PTOK
#define DIM     768
#define DHEAD   64
#define QKVDIM  2304          // 3*DIM
#define MROWS   (BATCH*NTOK)  // 25104 rows (divisible by 16 -> 1569 m-tiles)
#define MTILES  (MROWS/16)    // 1569
#define KPAD    224           // 7*32 K-padding for P@V wmma loop
#define KCHUNK  128           // B-panel k-chunk staged in LDS per GEMM block

typedef __attribute__((ext_vector_type(16))) __bf16          v16bf;
typedef __attribute__((ext_vector_type(8)))  float           v8f;
typedef __attribute__((ext_vector_type(8)))  unsigned short  u16x8;
typedef __attribute__((ext_vector_type(4)))  unsigned int    v4ui;
typedef __attribute__((ext_vector_type(8)))  int             v8ii;
typedef __attribute__((ext_vector_type(4)))  int             v4ii;

#if defined(__has_builtin)
#  if __has_builtin(__builtin_amdgcn_tensor_load_to_lds) && \
      __has_builtin(__builtin_amdgcn_s_wait_tensorcnt)
#    define HAVE_TDM 1
#  endif
#endif
#ifndef HAVE_TDM
#  define HAVE_TDM 0
#endif

// ---------------------------------------------------------------------------
// Scalar bf16 helpers (bit-level, RNE) — buffers held as unsigned short.
// ---------------------------------------------------------------------------
__device__ __forceinline__ unsigned short f32_to_bf16(float f) {
  unsigned int u = __float_as_uint(f);
  u += 0x7FFFu + ((u >> 16) & 1u);
  return (unsigned short)(u >> 16);
}
__device__ __forceinline__ float bf16_to_f32(unsigned short h) {
  return __uint_as_float(((unsigned int)h) << 16);
}

// ---------------------------------------------------------------------------
// WMMA fragment loader: 16x32 bf16 tile, row-major source.
//   lane&15 -> row, lane>>4 -> 8-elem K group.  Two 16B loads (b128).
// ---------------------------------------------------------------------------
__device__ __forceinline__ v16bf load_frag(const unsigned short* base, int ld) {
  int lane = threadIdx.x & 31;
  const unsigned short* p = base + (size_t)(lane & 15) * ld + ((lane >> 4) << 3);
  union { u16x8 h[2]; v16bf v; } u;
  u.h[0] = *(const u16x8*)p;
  u.h[1] = *(const u16x8*)(p + 16);
  return u.v;
}

__device__ __forceinline__ v8f wmma_bf16(v16bf a, v16bf b, v8f c) {
  return __builtin_amdgcn_wmma_f32_16x16x32_bf16(
      false, a, false, b, (short)0, c, false, false);
}

// ---------------------------------------------------------------------------
// Tensor Data Mover: 2-D tile (tile_x elems per row, tile_y rows, bf16) from
// global row-major (row_stride_elems) into LDS (contiguous, ld = tile_x).
// D# layout per CDNA5 ISA 8.3/8.4: group0 {count=1, lds_addr, global_addr,
// type=2}; group1 {data_size=2B, tensor dims, tile dims, stride0}.
// ---------------------------------------------------------------------------
__device__ __forceinline__ unsigned lds_byte_off(const void* p) {
  return (unsigned)(unsigned long long)p;   // flat addr low 32 bits = LDS offset
}

__device__ __forceinline__ void tdm_load_tile_2d(unsigned lds_addr,
                                                 const void* global_tile,
                                                 unsigned tile_x, unsigned tile_y,
                                                 unsigned row_stride_elems) {
#if HAVE_TDM
  unsigned long long ga = (unsigned long long)global_tile;
  const unsigned tdim0 = 0x100000u, tdim1 = 0x100000u;  // large: tile is interior
  v4ui g0;
  g0[0] = 1u;                                            // count=1, user desc
  g0[1] = lds_addr;
  g0[2] = (unsigned)ga;
  g0[3] = (unsigned)((ga >> 32) & 0x1FFFFFFu) | 0x80000000u;  // addr[56:32], type=2
  v8ii g1;
  g1[0] = (int)(1u << 16);                               // data_size=1 -> 2 bytes
  g1[1] = (int)((tdim0 & 0xFFFFu) << 16);                // tensor_dim0[15:0]
  g1[2] = (int)(((tdim0 >> 16) & 0xFFFFu) | ((tdim1 & 0xFFFFu) << 16));
  g1[3] = (int)(((tdim1 >> 16) & 0xFFFFu) | ((tile_x & 0xFFFFu) << 16));
  g1[4] = (int)(tile_y & 0xFFFFu);                       // tile_dim1 (tile_dim2=0)
  g1[5] = (int)row_stride_elems;                         // stride0[31:0]
  g1[6] = 0;
  g1[7] = 0;
  v4ii gz; gz[0] = 0; gz[1] = 0; gz[2] = 0; gz[3] = 0;   // 2-D: groups 2/3 null
#if __clang_major__ >= 23
  v8ii gz8; gz8[0]=0; gz8[1]=0; gz8[2]=0; gz8[3]=0; gz8[4]=0; gz8[5]=0; gz8[6]=0; gz8[7]=0;
  __builtin_amdgcn_tensor_load_to_lds(g0, g1, gz, gz, gz8, 0);
#else
  __builtin_amdgcn_tensor_load_to_lds(g0, g1, gz, gz, 0);
#endif
#else
  (void)lds_addr; (void)global_tile; (void)tile_x; (void)tile_y; (void)row_stride_elems;
#endif
}

// ---------------------------------------------------------------------------
// Kernel 1: elementwise fp32 -> bf16 cast
// ---------------------------------------------------------------------------
__global__ __launch_bounds__(256)
void cast_bf16_kernel(const float* __restrict__ in, unsigned short* __restrict__ out, int n) {
  int i = blockIdx.x * 256 + threadIdx.x;
  if (i < n) out[i] = f32_to_bf16(in[i]);
}

// ---------------------------------------------------------------------------
// Kernel 2: transpose + cast (R x C fp32) -> (C x R bf16)
// ---------------------------------------------------------------------------
__global__ __launch_bounds__(256)
void transpose_cast_kernel(const float* __restrict__ in, unsigned short* __restrict__ out,
                           int R, int C) {
  __shared__ float tile[16][17];
  int r0 = blockIdx.y * 16, c0 = blockIdx.x * 16;
  int tr = threadIdx.y, tc = threadIdx.x;
  if (r0 + tr < R && c0 + tc < C) tile[tr][tc] = in[(size_t)(r0 + tr) * C + (c0 + tc)];
  __syncthreads();
  if (c0 + tr < C && r0 + tc < R)
    out[(size_t)(c0 + tr) * R + (r0 + tc)] = f32_to_bf16(tile[tc][tr]);
}

// ---------------------------------------------------------------------------
// Shared GEMM core pattern (QKV + proj):
//   block = 256 threads (8 waves); block tile = 32 rows/wave-pair... each wave
//   owns 2 m-tiles (32 rows) x 64 cols -> 8 accumulators; block covers 16
//   m-tiles.  B-panel (64 cols x 128 k, bf16, 16 KB) staged in LDS, double
//   buffered, via TDM issued by wave 0 one chunk ahead (fallback: coop copy).
//   A-fragments from global, software-pipelined one k-step ahead.
// ---------------------------------------------------------------------------
#define GEMM_STAGE(dst_buf, chunk, wT)                                          \
  do {                                                                          \
    if (HAVE_TDM) {                                                             \
      if (threadIdx.x < 32)                                                     \
        tdm_load_tile_2d(lds_byte_off(&bpanel[dst_buf][0]),                     \
                         (wT) + (size_t)c0 * DIM + (size_t)(chunk) * KCHUNK,    \
                         KCHUNK, 64, DIM);                                      \
    } else {                                                                    \
      for (int idx = threadIdx.x; idx < 64 * (KCHUNK / 8); idx += 256) {        \
        int r = idx / (KCHUNK / 8), g8 = (idx % (KCHUNK / 8)) << 3;             \
        *(u16x8*)&bpanel[dst_buf][r * KCHUNK + g8] =                            \
            *(const u16x8*)&(wT)[(size_t)(c0 + r) * DIM + (size_t)(chunk) * KCHUNK + g8]; \
      }                                                                         \
    }                                                                           \
  } while (0)

#define GEMM_STAGE_WAIT()                                                       \
  do {                                                                          \
    if (HAVE_TDM) {                                                             \
      if (threadIdx.x < 32) {                                                   \
        __builtin_amdgcn_s_wait_tensorcnt(0);                                   \
      }                                                                         \
    }                                                                           \
    __syncthreads();                                                            \
  } while (0)

// ---------------------------------------------------------------------------
// Kernel 3: QKV GEMM.  Writes q (x0.125) / k / v head-major bf16.
// ---------------------------------------------------------------------------
__global__ __launch_bounds__(256)
void qkv_gemm_kernel(const unsigned short* __restrict__ xb,
                     const unsigned short* __restrict__ wqT,
                     unsigned short* __restrict__ qbf,
                     unsigned short* __restrict__ kbf,
                     unsigned short* __restrict__ vbf) {
  __shared__ unsigned short bpanel[2][64 * KCHUNK];   // 2 x 16 KB
  int wv = threadIdx.x >> 5, lane = threadIdx.x & 31;
  int mt0 = blockIdx.x * 16 + wv * 2;
  int mt1 = mt0 + 1;
  int mt0c = mt0 < MTILES ? mt0 : MTILES - 1;
  int mt1c = mt1 < MTILES ? mt1 : MTILES - 1;
  int c0 = blockIdx.y * 64;
  const unsigned short* a0base = xb + (size_t)mt0c * 16 * DIM;
  const unsigned short* a1base = xb + (size_t)mt1c * 16 * DIM;

  GEMM_STAGE(0, 0, wqT);
  GEMM_STAGE_WAIT();

  v8f acc[2][4];
#pragma unroll
  for (int m = 0; m < 2; m++)
#pragma unroll
    for (int t = 0; t < 4; t++) acc[m][t] = (v8f){};

  for (int c = 0; c < DIM / KCHUNK; c++) {
    if (c + 1 < DIM / KCHUNK) GEMM_STAGE((c + 1) & 1, c + 1, wqT);
    const unsigned short* bp = &bpanel[c & 1][0];
    v16bf A0 = load_frag(a0base + c * KCHUNK, DIM);
    v16bf A1 = load_frag(a1base + c * KCHUNK, DIM);
#pragma unroll
    for (int ks = 0; ks < KCHUNK / 32; ks++) {
      v16bf nA0, nA1;
      if (ks + 1 < KCHUNK / 32) {                 // pipeline next A k-step
        nA0 = load_frag(a0base + c * KCHUNK + (ks + 1) * 32, DIM);
        nA1 = load_frag(a1base + c * KCHUNK + (ks + 1) * 32, DIM);
      }
#pragma unroll
      for (int t = 0; t < 4; t++) {
        v16bf b = load_frag(bp + (size_t)t * 16 * KCHUNK + ks * 32, KCHUNK);
        acc[0][t] = wmma_bf16(A0, b, acc[0][t]);
        acc[1][t] = wmma_bf16(A1, b, acc[1][t]);
      }
      if (ks + 1 < KCHUNK / 32) { A0 = nA0; A1 = nA1; }
    }
    GEMM_STAGE_WAIT();
  }

  int colbase = lane & 15, rbase = (lane >> 4) * 8;
#pragma unroll
  for (int m = 0; m < 2; m++) {
    int mt = (m == 0) ? mt0 : mt1;
    if (mt >= MTILES) continue;
#pragma unroll
    for (int t = 0; t < 4; t++) {
      int cg = c0 + t * 16 + colbase;
      int which = cg / DIM;                // 0=q 1=k 2=v
      int within = cg % DIM;
      int head = within >> 6, dc = within & 63;
      unsigned short* dst = (which == 0) ? qbf : ((which == 1) ? kbf : vbf);
      float mul = (which == 0) ? 0.125f : 1.0f;
#pragma unroll
      for (int i = 0; i < 8; i++) {
        int g = mt * 16 + rbase + i;
        int bi = g / NTOK, tok = g % NTOK;
        dst[(((size_t)bi * HEADS + head) * NTOK + tok) * DHEAD + dc] =
            f32_to_bf16(acc[m][t][i] * mul);
      }
    }
  }
}

// ---------------------------------------------------------------------------
// Kernel 4: cls-token attention (tiny).  One block per (b,h).
// ---------------------------------------------------------------------------
__global__ __launch_bounds__(256)
void cls_attn_kernel(const unsigned short* __restrict__ qb,
                     const unsigned short* __restrict__ kb,
                     const unsigned short* __restrict__ vb,
                     const float* __restrict__ mask,
                     unsigned short* __restrict__ attb) {
  int bh = blockIdx.x;
  int b = bh / HEADS, h = bh % HEADS;
  int tid = threadIdx.x;
  __shared__ float sc[NTOK];
  __shared__ float qs[DHEAD];
  __shared__ float red[8];
  __shared__ float outred[4][DHEAD];

  const unsigned short* qrow = qb + (size_t)bh * NTOK * DHEAD;
  if (tid < DHEAD) qs[tid] = bf16_to_f32(qrow[tid]);
  __syncthreads();

  for (int j = tid; j < NTOK; j += 256) {
    const u16x8* kr8 = (const u16x8*)(kb + ((size_t)bh * NTOK + j) * DHEAD);
    float s = 0.f;
#pragma unroll
    for (int d8 = 0; d8 < DHEAD / 8; d8++) {
      u16x8 kv = kr8[d8];
#pragma unroll
      for (int e = 0; e < 8; e++) s += qs[d8 * 8 + e] * bf16_to_f32(kv[e]);
    }
    int mj = (j == 0) ? 0 : 1 + ((j - 1) % PTOK);
    s += (1.0f - mask[b * (1 + PTOK) + mj]) * -10000.0f;
    sc[j] = s;
  }
  __syncthreads();

  float mx = -3.0e38f;
  for (int j = tid; j < NTOK; j += 256) mx = fmaxf(mx, sc[j]);
  for (int m = 16; m >= 1; m >>= 1) mx = fmaxf(mx, __shfl_xor(mx, m, 32));
  if ((tid & 31) == 0) red[tid >> 5] = mx;
  __syncthreads();
  mx = red[0];
  for (int w = 1; w < 8; w++) mx = fmaxf(mx, red[w]);
  __syncthreads();

  float sm = 0.f;
  for (int j = tid; j < NTOK; j += 256) { float p = __expf(sc[j] - mx); sc[j] = p; sm += p; }
  for (int m = 16; m >= 1; m >>= 1) sm += __shfl_xor(sm, m, 32);
  if ((tid & 31) == 0) red[tid >> 5] = sm;
  __syncthreads();
  sm = 0.f;
  for (int w = 0; w < 8; w++) sm += red[w];
  float rinv = 1.0f / sm;

  int dd = tid & 63, g = tid >> 6;
  float acc = 0.f;
  for (int j = g; j < NTOK; j += 4)
    acc += sc[j] * bf16_to_f32(vb[((size_t)bh * NTOK + j) * DHEAD + dd]);
  outred[g][dd] = acc;
  __syncthreads();
  if (g == 0) {
    float tot = outred[0][dd] + outred[1][dd] + outred[2][dd] + outred[3][dd];
    attb[(size_t)b * NTOK * DIM + (size_t)h * DHEAD + dd] = f32_to_bf16(tot * rinv);
  }
}

// ---------------------------------------------------------------------------
// Kernel 5: spatial attention (WMMA).  One block (4 waves) per (b,h,f).
// ---------------------------------------------------------------------------
__device__ __forceinline__ v16bf load_frag_qsp(const unsigned short* qtile0, int mt, int k0) {
  int lane = threadIdx.x & 31;
  int qr = mt * 16 + (lane & 15);
  if (qr > PTOK - 1) qr = PTOK - 1;
  const unsigned short* p = qtile0 + (size_t)qr * DHEAD + k0 + ((lane >> 4) << 3);
  union { u16x8 h[2]; v16bf v; } u;
  u.h[0] = *(const u16x8*)p;
  u.h[1] = *(const u16x8*)(p + 16);
  return u.v;
}
__device__ __forceinline__ v16bf load_frag_ksp(const unsigned short* kbh, int fi, int t, int k0) {
  int lane = threadIdx.x & 31;
  int j = t * 16 + (lane & 15);
  int tk = (j == 0) ? 0 : ((j <= PTOK) ? (1 + fi * PTOK + (j - 1)) : 0);
  const unsigned short* p = kbh + (size_t)tk * DHEAD + k0 + ((lane >> 4) << 3);
  union { u16x8 h[2]; v16bf v; } u;
  u.h[0] = *(const u16x8*)p;
  u.h[1] = *(const u16x8*)(p + 16);
  return u.v;
}

__global__ __launch_bounds__(128)
void spatial_attn_kernel(const unsigned short* __restrict__ qb,
                         const unsigned short* __restrict__ kb,
                         const unsigned short* __restrict__ vb,
                         const float* __restrict__ mask,
                         unsigned short* __restrict__ attb) {
  int bhf = blockIdx.x;
  int fi = bhf % FRAMES;
  int bh = bhf / FRAMES;
  int b = bh / HEADS, h = bh % HEADS;
  int tid = threadIdx.x, lane = tid & 31, wv = tid >> 5;

  __shared__ unsigned short sh_vT[DHEAD * KPAD];   // V2^T [d][key], 28 KB
  __shared__ unsigned short sh_p[4][16 * KPAD];    // per-wave P scratch, 28 KB

  const unsigned short* qbh = qb + (size_t)bh * NTOK * DHEAD;
  const unsigned short* kbh = kb + (size_t)bh * NTOK * DHEAD;
  const unsigned short* vbh = vb + (size_t)bh * NTOK * DHEAD;
  const unsigned short* qtile0 = qbh + (size_t)(1 + fi * PTOK) * DHEAD;

  for (int r = tid; r < KPAD; r += 128) {
    if (r <= PTOK) {
      int tk = (r == 0) ? 0 : (1 + fi * PTOK + (r - 1));
      const unsigned short* vr = vbh + (size_t)tk * DHEAD;
      for (int d2 = 0; d2 < DHEAD; d2++) sh_vT[d2 * KPAD + r] = vr[d2];
    } else {
      for (int d2 = 0; d2 < DHEAD; d2++) sh_vT[d2 * KPAD + r] = 0;
    }
  }
  for (int i = tid; i < 4 * 16 * KPAD; i += 128) ((unsigned short*)sh_p)[i] = 0;
  __syncthreads();

  int colbase = lane & 15, rbase = (lane >> 4) * 8;
  for (int mt = wv; mt < 13; mt += 4) {
    v16bf aq0 = load_frag_qsp(qtile0, mt, 0);
    v16bf aq1 = load_frag_qsp(qtile0, mt, 32);
    v8f st[13];
#pragma unroll
    for (int t = 0; t < 13; t++) {
      v8f c = {};
      c = wmma_bf16(aq0, load_frag_ksp(kbh, fi, t, 0), c);
      c = wmma_bf16(aq1, load_frag_ksp(kbh, fi, t, 32), c);
      st[t] = c;
    }
    float mx[8], sm[8];
#pragma unroll
    for (int i = 0; i < 8; i++) mx[i] = -3.0e38f;
#pragma unroll
    for (int t = 0; t < 13; t++) {
      int j = t * 16 + colbase;
      float madd = (j <= PTOK) ? (1.0f - mask[b * (1 + PTOK) + j]) * -10000.0f
                               : -3.0e38f;
#pragma unroll
      for (int i = 0; i < 8; i++) {
        float s = st[t][i] + madd;
        st[t][i] = s;
        mx[i] = fmaxf(mx[i], s);
      }
    }
#pragma unroll
    for (int i = 0; i < 8; i++) {
      float m = mx[i];
      m = fmaxf(m, __shfl_xor(m, 1, 32));
      m = fmaxf(m, __shfl_xor(m, 2, 32));
      m = fmaxf(m, __shfl_xor(m, 4, 32));
      m = fmaxf(m, __shfl_xor(m, 8, 32));
      mx[i] = m;
    }
#pragma unroll
    for (int i = 0; i < 8; i++) sm[i] = 0.f;
    unsigned short* pw = &sh_p[wv][0];
#pragma unroll
    for (int t = 0; t < 13; t++) {
#pragma unroll
      for (int i = 0; i < 8; i++) {
        float p = __expf(st[t][i] - mx[i]);
        sm[i] += p;
        pw[(size_t)(rbase + i) * KPAD + t * 16 + colbase] = f32_to_bf16(p);
      }
    }
#pragma unroll
    for (int i = 0; i < 8; i++) {
      float s = sm[i];
      s += __shfl_xor(s, 1, 32);
      s += __shfl_xor(s, 2, 32);
      s += __shfl_xor(s, 4, 32);
      s += __shfl_xor(s, 8, 32);
      sm[i] = 1.0f / s;
    }
    v8f o0 = {}, o1 = {}, o2 = {}, o3 = {};
#pragma unroll
    for (int ks = 0; ks < 7; ks++) {
      v16bf ap = load_frag(pw + ks * 32, KPAD);
      o0 = wmma_bf16(ap, load_frag(sh_vT + (size_t)( 0) * KPAD + ks * 32, KPAD), o0);
      o1 = wmma_bf16(ap, load_frag(sh_vT + (size_t)(16) * KPAD + ks * 32, KPAD), o1);
      o2 = wmma_bf16(ap, load_frag(sh_vT + (size_t)(32) * KPAD + ks * 32, KPAD), o2);
      o3 = wmma_bf16(ap, load_frag(sh_vT + (size_t)(48) * KPAD + ks * 32, KPAD), o3);
    }
    unsigned short* ob = attb + ((size_t)b * NTOK + 1 + (size_t)fi * PTOK) * DIM
                        + (size_t)h * DHEAD;
    v8f oo[4] = {o0, o1, o2, o3};
#pragma unroll
    for (int nd = 0; nd < 4; nd++) {
#pragma unroll
      for (int i = 0; i < 8; i++) {
        int qr = mt * 16 + rbase + i;
        if (qr < PTOK)
          ob[(size_t)qr * DIM + nd * 16 + colbase] = f32_to_bf16(oo[nd][i] * sm[i]);
      }
    }
  }
}

// ---------------------------------------------------------------------------
// Kernel 6: projection GEMM + bias -> fp32 output (same structure as QKV).
// ---------------------------------------------------------------------------
__global__ __launch_bounds__(256)
void proj_gemm_kernel(const unsigned short* __restrict__ attb,
                      const unsigned short* __restrict__ wpT,
                      const float* __restrict__ bproj,
                      float* __restrict__ out) {
  __shared__ unsigned short bpanel[2][64 * KCHUNK];
  int wv = threadIdx.x >> 5, lane = threadIdx.x & 31;
  int mt0 = blockIdx.x * 16 + wv * 2;
  int mt1 = mt0 + 1;
  int mt0c = mt0 < MTILES ? mt0 : MTILES - 1;
  int mt1c = mt1 < MTILES ? mt1 : MTILES - 1;
  int c0 = blockIdx.y * 64;
  const unsigned short* a0base = attb + (size_t)mt0c * 16 * DIM;
  const unsigned short* a1base = attb + (size_t)mt1c * 16 * DIM;

  GEMM_STAGE(0, 0, wpT);
  GEMM_STAGE_WAIT();

  v8f acc[2][4];
#pragma unroll
  for (int m = 0; m < 2; m++)
#pragma unroll
    for (int t = 0; t < 4; t++) acc[m][t] = (v8f){};

  for (int c = 0; c < DIM / KCHUNK; c++) {
    if (c + 1 < DIM / KCHUNK) GEMM_STAGE((c + 1) & 1, c + 1, wpT);
    const unsigned short* bp = &bpanel[c & 1][0];
    v16bf A0 = load_frag(a0base + c * KCHUNK, DIM);
    v16bf A1 = load_frag(a1base + c * KCHUNK, DIM);
#pragma unroll
    for (int ks = 0; ks < KCHUNK / 32; ks++) {
      v16bf nA0, nA1;
      if (ks + 1 < KCHUNK / 32) {
        nA0 = load_frag(a0base + c * KCHUNK + (ks + 1) * 32, DIM);
        nA1 = load_frag(a1base + c * KCHUNK + (ks + 1) * 32, DIM);
      }
#pragma unroll
      for (int t = 0; t < 4; t++) {
        v16bf b = load_frag(bp + (size_t)t * 16 * KCHUNK + ks * 32, KCHUNK);
        acc[0][t] = wmma_bf16(A0, b, acc[0][t]);
        acc[1][t] = wmma_bf16(A1, b, acc[1][t]);
      }
      if (ks + 1 < KCHUNK / 32) { A0 = nA0; A1 = nA1; }
    }
    GEMM_STAGE_WAIT();
  }

  int colbase = lane & 15, rbase = (lane >> 4) * 8;
#pragma unroll
  for (int m = 0; m < 2; m++) {
    int mt = (m == 0) ? mt0 : mt1;
    if (mt >= MTILES) continue;
#pragma unroll
    for (int t = 0; t < 4; t++) {
      int cg = c0 + t * 16 + colbase;
      float bias = bproj[cg];
#pragma unroll
      for (int i = 0; i < 8; i++) {
        int g = mt * 16 + rbase + i;
        out[(size_t)g * DIM + cg] = acc[m][t][i] + bias;
      }
    }
  }
}

// ---------------------------------------------------------------------------
// Host launch
// ---------------------------------------------------------------------------
extern "C" void kernel_launch(void* const* d_in, const int* in_sizes, int n_in,
                              void* d_out, int out_size, void* d_ws, size_t ws_size,
                              hipStream_t stream) {
  (void)in_sizes; (void)n_in; (void)out_size; (void)ws_size;
  const float* x     = (const float*)d_in[0];
  const float* mask  = (const float*)d_in[1];
  const float* Wqkv  = (const float*)d_in[2];
  const float* Wproj = (const float*)d_in[3];
  const float* bproj = (const float*)d_in[4];
  float* out = (float*)d_out;

  char* ws = (char*)d_ws;
  size_t off = 0;
  auto alloc = [&](size_t bytes) -> void* {
    void* p = ws + off;
    off += (bytes + 255) & ~(size_t)255;
    return p;
  };
  unsigned short* xb   = (unsigned short*)alloc((size_t)MROWS * DIM * 2);
  unsigned short* wqT  = (unsigned short*)alloc((size_t)QKVDIM * DIM * 2);
  unsigned short* wpT  = (unsigned short*)alloc((size_t)DIM * DIM * 2);
  unsigned short* qbf  = (unsigned short*)alloc((size_t)BATCH*HEADS*NTOK*DHEAD*2);
  unsigned short* kbf  = (unsigned short*)alloc((size_t)BATCH*HEADS*NTOK*DHEAD*2);
  unsigned short* vbf  = (unsigned short*)alloc((size_t)BATCH*HEADS*NTOK*DHEAD*2);
  unsigned short* attb = (unsigned short*)alloc((size_t)MROWS * DIM * 2);

  {
    int n = MROWS * DIM;
    cast_bf16_kernel<<<(n + 255) / 256, 256, 0, stream>>>(x, xb, n);
  }
  {
    dim3 blk(16, 16);
    dim3 g1(QKVDIM / 16, DIM / 16);
    transpose_cast_kernel<<<g1, blk, 0, stream>>>(Wqkv, wqT, DIM, QKVDIM);
    dim3 g2(DIM / 16, DIM / 16);
    transpose_cast_kernel<<<g2, blk, 0, stream>>>(Wproj, wpT, DIM, DIM);
  }
  {
    dim3 grid((MTILES + 15) / 16, QKVDIM / 64);
    qkv_gemm_kernel<<<grid, 256, 0, stream>>>(xb, wqT, qbf, kbf, vbf);
  }
  cls_attn_kernel<<<BATCH * HEADS, 256, 0, stream>>>(qbf, kbf, vbf, mask, attb);
  spatial_attn_kernel<<<BATCH * HEADS * FRAMES, 128, 0, stream>>>(qbf, kbf, vbf, mask, attb);
  {
    dim3 grid((MTILES + 15) / 16, DIM / 64);
    proj_gemm_kernel<<<grid, 256, 0, stream>>>(attb, wpT, bproj, out);
  }
}
